// ASPBlock_72619307041051
// MI455X (gfx1250) — matrix operand order repaired and checked by reading, not compiled, not run on hardware
//
#include <hip/hip_runtime.h>
#include <math.h>

typedef __attribute__((ext_vector_type(16))) __bf16 v16bf;
typedef __attribute__((ext_vector_type(8)))  float  v8f;
typedef __attribute__((ext_vector_type(4)))  int    v4i_;

#define Bn 32
#define Cn 1536
#define Tn 800
#define An 128
#define C3 4608   // 3*Cn
#define TW 80     // t-columns per gemm1 block = 5 WMMA n-tiles

#if defined(__has_builtin)
#if __has_builtin(__builtin_amdgcn_global_load_async_to_lds_b128)
#define HAVE_ASYNC_LDS 1
#endif
#endif

typedef __attribute__((address_space(1))) v4i_* gv4i_p;
typedef __attribute__((address_space(3))) v4i_* lv4i_p;

__device__ __forceinline__ unsigned short f2bf(float f) {
  unsigned u = __float_as_uint(f);
  u += 0x7fffu + ((u >> 16) & 1u);          // round-to-nearest-even
  return (unsigned short)(u >> 16);
}

// ---------------- Kernel 1: per-(b,c) mean / std over T -------------------
__global__ __launch_bounds__(256)
void aspb_stats(const float* __restrict__ x, float* __restrict__ mean,
                float* __restrict__ sd) {
  __shared__ float s1[8], s2[8];
  const int row = blockIdx.x;                   // b*Cn + c
  const float* xr = x + (size_t)row * Tn;
  float s = 0.f, ss = 0.f;
  for (int t = threadIdx.x; t < Tn; t += 256) {
    float v = xr[t];
    s += v; ss += v * v;
  }
  #pragma unroll
  for (int m = 16; m >= 1; m >>= 1) {
    s  += __shfl_xor(s,  m, 32);
    ss += __shfl_xor(ss, m, 32);
  }
  if ((threadIdx.x & 31) == 0) { s1[threadIdx.x >> 5] = s; s2[threadIdx.x >> 5] = ss; }
  __syncthreads();
  if (threadIdx.x == 0) {
    float S = 0.f, SS = 0.f;
    #pragma unroll
    for (int i = 0; i < 8; ++i) { S += s1[i]; SS += s2[i]; }
    float mu  = S / (float)Tn;
    float var = (SS - S * mu) / (float)(Tn - 1);     // ddof=1
    mean[row] = mu;
    sd[row]   = sqrtf(fmaxf(var, 1e-4f));
  }
}

// ------ Kernel 2: fold W1[:,C:3C] @ [mu;sd] + b1 -> per-(b,a) bias --------
__global__ __launch_bounds__(128)
void aspb_hbias(const float* __restrict__ W1, const float* __restrict__ b1,
                const float* __restrict__ mean, const float* __restrict__ sd,
                float* __restrict__ hb) {
  const int b = blockIdx.x, a = threadIdx.x;
  const float* wr = W1 + (size_t)a * C3;
  const float* mu = mean + (size_t)b * Cn;
  const float* sg = sd   + (size_t)b * Cn;
  float acc = b1[a];
  #pragma unroll 4
  for (int c = 0; c < Cn; ++c)
    acc += wr[Cn + c] * mu[c] + wr[2 * Cn + c] * sg[c];
  hb[b * An + a] = acc;
}

// -------- Kernel 2b: one-time fp32 -> bf16 conversion of W1a and W2 -------
__global__ __launch_bounds__(256)
void aspb_prep(const float* __restrict__ W1, const float* __restrict__ W2,
               unsigned short* __restrict__ W1b, unsigned short* __restrict__ W2b) {
  int e = blockIdx.x * 256 + threadIdx.x;       // 0 .. 2*128*1536-1
  if (e < An * Cn) {
    int a = e / Cn, k = e - a * Cn;
    W1b[e] = f2bf(W1[(size_t)a * C3 + k]);      // only the x-block of W1
  } else {
    int i = e - An * Cn;
    W2b[i] = f2bf(W2[i]);                       // W2 is [Cn][An] row-major
  }
}

// ---- Kernel 3: H = tanh(bn1(relu(W1a @ x + hbias)))  [WMMA bf16] ---------
// Block = (b, 80 t-columns). 8 waves x 16-row m-tiles, 5 n-tiles each.
// Writes H t-major: Ht[b][t][a] (packed bf16) for straight staging in K4.
__global__ __launch_bounds__(256)
void aspb_gemm1(const float* __restrict__ x,
                const unsigned short* __restrict__ W1b,
                const float* __restrict__ hb,
                const float* __restrict__ g1, const float* __restrict__ be1,
                const float* __restrict__ mn1, const float* __restrict__ vr1,
                unsigned short* __restrict__ Ht) {
  __shared__ unsigned short lA[An * 32];        // [a 0..127][k 0..31] bf16
  __shared__ unsigned short lB[5 * 16 * 32];    // [nt][t 0..15][k 0..31] bf16
  const int b    = blockIdx.x;
  const int t0   = blockIdx.y * TW;
  const int tid  = threadIdx.x;
  const int lane = tid & 31, wave = tid >> 5;

  v8f zero = {};
  v8f acc[5];
  #pragma unroll
  for (int i = 0; i < 5; ++i) acc[i] = zero;

  const unsigned* W1u = (const unsigned*)W1b;   // bf16 pairs
  const float* xb = x + (size_t)b * Cn * Tn;
  unsigned* uA = (unsigned*)lA;
  const unsigned* pA = (const unsigned*)lA;
  const unsigned* pB = (const unsigned*)lB;
  const int rA  = (wave << 4) + (lane & 15);    // A-matrix row (channel a)
  const int kbA = (lane >> 4) << 4;             // 0 or 16
  const int tB  = lane & 15;                    // B column within n-tile
  const int kbB = (lane >> 4) << 4;             // 0 or 16

  for (int k0 = 0; k0 < Cn; k0 += 32) {
    // -- staging pass 1: issue all global loads into temp regs (one clause) --
    unsigned ta[8];
    #pragma unroll
    for (int i = 0; i < 8; ++i) {               // 2048 uints of W1b tile
      int e = tid + (i << 8);
      ta[i] = W1u[(size_t)(e >> 4) * (Cn / 2) + (k0 >> 1) + (e & 15)];
    }
    float tb[10];
    #pragma unroll
    for (int i = 0; i < 10; ++i) {              // 2560 floats of x tile
      int e = tid + (i << 8);
      int r = e / TW;                           // k row 0..31
      int tt = e - r * TW;                      // 0..79
      tb[i] = xb[(size_t)(k0 + r) * Tn + t0 + tt];
    }
    if (k0 + 32 < Cn)                           // prefetch next x tile
      __builtin_prefetch(&xb[(size_t)(k0 + 32 + (tid & 31)) * Tn +
                             t0 + ((tid >> 5) << 3)], 0, 0);
    // -- staging pass 2: convert + LDS stores --
    #pragma unroll
    for (int i = 0; i < 8; ++i) {
      int e = tid + (i << 8);
      uA[e] = ta[i];
    }
    #pragma unroll
    for (int i = 0; i < 10; ++i) {
      int e = tid + (i << 8);
      int r = e / TW;
      int tt = e - r * TW;
      lB[((tt >> 4) << 9) + ((tt & 15) << 5) + r] = f2bf(tb[i]);
    }
    __syncthreads();
    union { unsigned u[8]; v16bf v; } fa;
    #pragma unroll
    for (int v = 0; v < 8; ++v) {
      int K = kbA + (v << 1);                            // contiguous k = 16h + s, same order as B
      fa.u[v] = pA[((rA << 5) + K) >> 1];
    }
    #pragma unroll
    for (int nt = 0; nt < 5; ++nt) {
      union { unsigned u[8]; v16bf v; } fb;
      #pragma unroll
      for (int v = 0; v < 8; ++v) {
        int K2 = kbB + (v << 1);                         // contiguous k = 16h + s, same order as A
        fb.u[v] = pB[((nt << 9) + (tB << 5) + K2) >> 1];
      }
      acc[nt] = __builtin_amdgcn_wmma_f32_16x16x32_bf16(false, fa.v, false, fb.v,
                                                        (short)0, acc[nt], false, false);
    }
    __syncthreads();
  }

  const int mbase = (wave << 4) + ((lane >> 4) << 3);    // C-layout rows
  float sc[8], sh[8], hv[8];
  #pragma unroll
  for (int v = 0; v < 8; ++v) {
    int ch = mbase + v;
    float t = g1[ch] * rsqrtf(vr1[ch] + 1e-5f);
    sc[v] = t; sh[v] = be1[ch] - mn1[ch] * t;
    hv[v] = hb[b * An + ch];
  }
  #pragma unroll
  for (int nt = 0; nt < 5; ++nt) {
    const int t = t0 + (nt << 4) + (lane & 15);
    unsigned op[4];
    #pragma unroll
    for (int v = 0; v < 8; ++v) {
      float h = acc[nt][v] + hv[v];
      h = fmaxf(h, 0.f);
      h = tanhf(h * sc[v] + sh[v]);
      unsigned short hbv = f2bf(h);
      if (v & 1) op[v >> 1] |= ((unsigned)hbv) << 16;
      else       op[v >> 1]  = (unsigned)hbv;
    }
    uint4 o4; o4.x = op[0]; o4.y = op[1]; o4.z = op[2]; o4.w = op[3];
    *(uint4*)(Ht + ((size_t)b * Tn + t) * An + mbase) = o4;   // 16B store
  }
}

// ---- Kernel 4: a = bn2(relu(W2 @ H + b2)); fused softmax+pooling ---------
// One block = (b, 16 output channels). All of H[b] (200KB bf16) lives in LDS.
__global__ __launch_bounds__(256)
void aspb_gemm2_pool(const float* __restrict__ x,
                     const unsigned short* __restrict__ Ht,
                     const unsigned short* __restrict__ W2b,
                     const float* __restrict__ b2,
                     const float* __restrict__ g2, const float* __restrict__ be2,
                     const float* __restrict__ mn2, const float* __restrict__ vr2,
                     float* __restrict__ out) {
  extern __shared__ unsigned char smem[];
  unsigned short* lH = (unsigned short*)smem;              // [800][128] bf16
  unsigned short* lA = (unsigned short*)(smem + 204800);   // [16][128] bf16
  float*          red = (float*)(smem + 204800 + 4096);    // [8 waves][16][4]

  const int b    = blockIdx.x;
  const int o0   = blockIdx.y << 4;
  const int tid  = threadIdx.x, lane = tid & 31, wave = tid >> 5;

  { // stage H[b] -> LDS (t-major already: straight 16B-granule copy)
    const uint4* src = (const uint4*)(Ht + (size_t)b * Tn * An);
    uint4* dst = (uint4*)lH;
#if defined(HAVE_ASYNC_LDS)
    for (int i = tid; i < (Tn * An / 8); i += 256)
      __builtin_amdgcn_global_load_async_to_lds_b128(
          (gv4i_p)(src + i), (lv4i_p)(dst + i), 0, 0);
    asm volatile("s_wait_asynccnt 0x0" ::: "memory");
#else
    for (int i = tid; i < (Tn * An / 8); i += 256) dst[i] = src[i];
#endif
  }
  { // stage W2b tile: rows o0..o0+15 are 1024 contiguous uints
    const unsigned* W2u = (const unsigned*)W2b;
    unsigned* uA = (unsigned*)lA;
    for (int i = tid; i < 1024; i += 256) uA[i] = W2u[((size_t)o0 << 6) + i];
  }
  __syncthreads();

  const unsigned* pA = (const unsigned*)lA;
  const unsigned* pH = (const unsigned*)lH;
  const int rA   = lane & 15;
  const int kbA  = (lane >> 4) << 4;
  const int tcol = lane & 15;
  const int kbB  = (lane >> 4) << 4;

  float sc[8], sf[8], bb[8];
  const float* xr[8];
  float m[8], s[8], wx[8], wxx[8];
  #pragma unroll
  for (int v = 0; v < 8; ++v) {
    int o = o0 + ((lane >> 4) << 3) + v;
    float t = g2[o] * rsqrtf(vr2[o] + 1e-5f);
    sc[v] = t; sf[v] = be2[o] - mn2[o] * t; bb[v] = b2[o];
    xr[v] = x + ((size_t)b * Cn + o) * Tn;
    m[v] = -INFINITY; s[v] = 0.f; wx[v] = 0.f; wxx[v] = 0.f;
  }

  for (int nt = wave; nt < Tn / 16; nt += 8) {       // waves sweep t-tiles
    const int t0 = nt << 4;
    v8f acc = {};
    #pragma unroll
    for (int ks = 0; ks < 4; ++ks) {                 // K = 128 = 4 x 32
      union { unsigned u[8]; v16bf v; } fa, fb;
      const int kb = ks << 5;
      #pragma unroll
      for (int v = 0; v < 8; ++v) {
        int K  = kb + kbA + (v << 1);
        fa.u[v] = pA[((rA << 7) + K) >> 1];
        int K2 = kb + kbB + (v << 1);
        fb.u[v] = pH[(((t0 + tcol) << 7) + K2) >> 1];
      }
      acc = __builtin_amdgcn_wmma_f32_16x16x32_bf16(false, fa.v, false, fb.v,
                                                    (short)0, acc, false, false);
    }
    const int t = t0 + tcol;
    #pragma unroll
    for (int v = 0; v < 8; ++v) {                    // online softmax+pool
      float a = acc[v] + bb[v];
      a = fmaxf(a, 0.f);
      a = a * sc[v] + sf[v];
      float xv = xr[v][t];
      float mn = fmaxf(m[v], a);
      float c1 = __expf(m[v] - mn);
      float c2 = __expf(a - mn);
      s[v]   = s[v]   * c1 + c2;
      wx[v]  = wx[v]  * c1 + c2 * xv;
      wxx[v] = wxx[v] * c1 + c2 * xv * xv;
      m[v] = mn;
    }
  }

  // merge across the 16 lanes of each half-wave (log-sum-exp combine)
  #pragma unroll
  for (int msk = 1; msk < 16; msk <<= 1) {
    #pragma unroll
    for (int v = 0; v < 8; ++v) {
      float om = __shfl_xor(m[v],   msk, 32);
      float os = __shfl_xor(s[v],   msk, 32);
      float oa = __shfl_xor(wx[v],  msk, 32);
      float ob = __shfl_xor(wxx[v], msk, 32);
      float mn = fmaxf(m[v], om);
      float c1 = __expf(m[v] - mn), c2 = __expf(om - mn);
      s[v]   = s[v]   * c1 + os * c2;
      wx[v]  = wx[v]  * c1 + oa * c2;
      wxx[v] = wxx[v] * c1 + ob * c2;
      m[v] = mn;
    }
  }
  if ((lane & 15) == 0) {
    int rb = (lane >> 4) << 3;
    #pragma unroll
    for (int v = 0; v < 8; ++v) {
      float* p = red + ((wave << 4) + rb + v) * 4;
      p[0] = m[v]; p[1] = s[v]; p[2] = wx[v]; p[3] = wxx[v];
    }
  }
  __syncthreads();
  if (tid < 16) {                                    // merge the 8 waves
    float M = -INFINITY, S = 0.f, WX = 0.f, WXX = 0.f;
    for (int w = 0; w < 8; ++w) {
      const float* p = red + ((w << 4) + tid) * 4;
      float mn = fmaxf(M, p[0]);
      float c1 = __expf(M - mn), c2 = __expf(p[0] - mn);
      S   = S   * c1 + p[1] * c2;
      WX  = WX  * c1 + p[2] * c2;
      WXX = WXX * c1 + p[3] * c2;
      M = mn;
    }
    float mu   = WX / S;
    float varw = WXX / S - mu * mu;
    float sg   = sqrtf(fmaxf(varw, 1e-4f));
    int o = o0 + tid;
    out[(size_t)b * (2 * Cn) + o]      = mu;
    out[(size_t)b * (2 * Cn) + Cn + o] = sg;
  }
}

extern "C" void kernel_launch(void* const* d_in, const int* in_sizes, int n_in,
                              void* d_out, int out_size, void* d_ws, size_t ws_size,
                              hipStream_t stream) {
  (void)in_sizes; (void)n_in; (void)out_size; (void)ws_size;
  const float* x   = (const float*)d_in[0];
  const float* W1  = (const float*)d_in[1];
  const float* b1  = (const float*)d_in[2];
  const float* g1  = (const float*)d_in[3];
  const float* be1 = (const float*)d_in[4];
  const float* mn1 = (const float*)d_in[5];
  const float* vr1 = (const float*)d_in[6];
  const float* W2  = (const float*)d_in[7];
  const float* b2  = (const float*)d_in[8];
  const float* g2  = (const float*)d_in[9];
  const float* be2 = (const float*)d_in[10];
  const float* mn2 = (const float*)d_in[11];
  const float* vr2 = (const float*)d_in[12];
  float* out = (float*)d_out;

  // workspace: mean/sd [B*C] f32, hbias [B*A] f32, W1b/W2b bf16, Ht bf16
  float* mean = (float*)d_ws;
  float* sd   = mean + (size_t)Bn * Cn;
  float* hbuf = sd   + (size_t)Bn * Cn;
  unsigned short* W1b = (unsigned short*)(hbuf + (size_t)Bn * An);
  unsigned short* W2b = W1b + (size_t)An * Cn;
  unsigned short* Ht  = W2b + (size_t)An * Cn;   // 16B-aligned offset

  aspb_stats<<<dim3(Bn * Cn), 256, 0, stream>>>(x, mean, sd);
  aspb_hbias<<<dim3(Bn), 128, 0, stream>>>(W1, b1, mean, sd, hbuf);
  aspb_prep<<<dim3(2 * An * Cn / 256), 256, 0, stream>>>(W1, W2, W1b, W2b);
  aspb_gemm1<<<dim3(Bn, Tn / TW), 256, 0, stream>>>(x, W1b, hbuf, g1, be1, mn1, vr1, Ht);

  const size_t smem = 204800 + 4096 + 2048;   // H[b] + W2 tile + reduce scratch
  (void)hipFuncSetAttribute((const void*)aspb_gemm2_pool,
                            hipFuncAttributeMaxDynamicSharedMemorySize, (int)smem);
  aspb_gemm2_pool<<<dim3(Bn, Cn / 16), 256, smem, stream>>>(
      x, Ht, W2b, b2, g2, be2, mn2, vr2, out);
}
